// GraphLayer_19851338842426
// MI455X (gfx1250) — matrix operand order repaired and checked by reading, not compiled, not run on hardware
//
#include <hip/hip_runtime.h>

typedef __attribute__((ext_vector_type(16))) __bf16 v16bf;
typedef __attribute__((ext_vector_type(4)))  __bf16 v4bf;
typedef __attribute__((ext_vector_type(8)))  float  v8f;

// Swizzle a K/F index into the CDNA5 16-bit A-fragment LDS order so that one
// lane's 16 fragment elements (K = kk + {0..7|16..23} for half 0, kk + {8..15|24..31}
// for half 1) are 32 contiguous bytes in LDS.
__device__ __forceinline__ int swizA(int k) {
    return (((k >> 5) << 1) + ((k >> 3) & 1)) * 16 + ((k >> 4) & 1) * 8 + (k & 7);
}

__global__ __launch_bounds__(256)
void graph_layer_fused(const float* __restrict__ x,
                       const float* __restrict__ adj,
                       const float* __restrict__ W,
                       const float* __restrict__ bias,
                       float* __restrict__ out)
{
    // Stage1: ldsA = adj tile (A-swizzled), ldsB = x tile (row-major K x F)
    // Stage2: ldsA = x_agg   (A-swizzled), ldsB = W^T    (row-major F x O)
    __shared__ __bf16 ldsA[128 * 128];
    __shared__ __bf16 ldsB[128 * 128];
    __shared__ float  ldsBias[128];

    const int tid  = threadIdx.x;
    const int wv   = tid >> 5;     // wave id 0..7  -> 16-row strip
    const int lane = tid & 31;
    const int hf   = lane >> 4;    // half-wave (K-split for A/B fragments)
    const int r    = lane & 15;    // row within A frag / col within B,C,D frags

    const int wg   = blockIdx.x;   // 0..3071
    const int nblk = wg & 3;       // 4 blocks of 128 rows over N=512
    const int bt   = wg >> 2;      // 0..767
    const int t    = bt % 48;
    const int b    = bt / 48;

    const float* adjB = adj + (size_t)b * 512 * 512 + (size_t)nblk * 128 * 512;
    const float* xBT  = x   + ((size_t)b * 48 + t) * 512 * 128;
    float*       oBT  = out + ((size_t)b * 48 + t) * 512 * 128 + (size_t)nblk * 128 * 128;

    if (tid < 128) ldsBias[tid] = bias[tid];

    const v8f vzero = {0.f, 0.f, 0.f, 0.f, 0.f, 0.f, 0.f, 0.f};
    v8f acc[8];
#pragma unroll
    for (int i = 0; i < 8; ++i) acc[i] = vzero;

    // ---------------- Stage 1: x_agg = adj[b] @ x[b,t] ----------------
    for (int mc = 0; mc < 512; mc += 128) {
        __syncthreads();   // previous iteration's fragment loads done before overwrite

        // adj tile: 128 n-rows x 128 k-cols, f32 -> bf16, A-fragment swizzle.
        // 4096 float4 / 256 threads = 16 each; 4 converted bf16 stay contiguous
        // under swizA (k0 % 4 == 0 keeps bits [3..6] fixed).
#pragma unroll 4
        for (int i = 0; i < 16; ++i) {
            int idx = tid + i * 256;
            int row = idx >> 5;
            int c4  = idx & 31;
            float4 v = *(const float4*)(adjB + (size_t)row * 512 + mc + c4 * 4);
            v4bf p = {(__bf16)v.x, (__bf16)v.y, (__bf16)v.z, (__bf16)v.w};
            *(v4bf*)&ldsA[row * 128 + swizA(c4 * 4)] = p;
        }
        // x tile: 128 k-rows x 128 f-cols, f32 -> bf16, plain row-major
        // (already matches the B-fragment read pattern).
#pragma unroll 4
        for (int i = 0; i < 16; ++i) {
            int idx = tid + i * 256;
            int row = idx >> 5;
            int c4  = idx & 31;
            float4 v = *(const float4*)(xBT + (size_t)(mc + row) * 128 + c4 * 4);
            v4bf p = {(__bf16)v.x, (__bf16)v.y, (__bf16)v.z, (__bf16)v.w};
            *(v4bf*)&ldsB[row * 128 + c4 * 4] = p;
        }
        __syncthreads();

#pragma unroll
        for (int kk = 0; kk < 128; kk += 32) {
            const v16bf a =
                *(const v16bf*)&ldsA[(wv * 16 + r) * 128 + (((kk >> 5) << 1) + hf) * 16];
#pragma unroll
            for (int nf = 0; nf < 8; ++nf) {
                v16bf bb;
#pragma unroll
                for (int s = 0; s < 16; ++s)
                    bb[s] = ldsB[(kk + (s & 8) * 2 + hf * 8 + (s & 7)) * 128 + nf * 16 + r];
                acc[nf] = __builtin_amdgcn_wmma_f32_16x16x32_bf16(
                    false, a, false, bb, (short)0, acc[nf], false, false);
            }
        }
    }

    __syncthreads();   // all waves done reading stage-1 tiles

    // ---------------- Stage 2 staging ----------------
    // x_agg (C/D layout: M = v + hf*8, N = r) -> bf16 A-fragment swizzle in ldsA
#pragma unroll
    for (int nf = 0; nf < 8; ++nf) {
#pragma unroll
        for (int v = 0; v < 8; ++v) {
            int row = wv * 16 + hf * 8 + v;
            int f   = nf * 16 + r;
            ldsA[row * 128 + swizA(f)] = (__bf16)acc[nf][v];
        }
    }
    // W^T: ldsB[f*128 + o] = W[o*128 + f] (coalesced global reads)
    for (int i = 0; i < 64; ++i) {
        int idx = tid + i * 256;
        int o = idx >> 7;
        int f = idx & 127;
        ldsB[f * 128 + o] = (__bf16)W[(size_t)o * 128 + f];
    }
    __syncthreads();

    // ---------------- Stage 2: out = x_agg @ W^T + bias ----------------
    v8f oacc[8];
#pragma unroll
    for (int nf = 0; nf < 8; ++nf) {
        float bb = ldsBias[nf * 16 + r];     // N = r in the C/D layout
        v8f s = {bb, bb, bb, bb, bb, bb, bb, bb};
        oacc[nf] = s;
    }
#pragma unroll
    for (int kk = 0; kk < 128; kk += 32) {
        const v16bf a =
            *(const v16bf*)&ldsA[(wv * 16 + r) * 128 + (((kk >> 5) << 1) + hf) * 16];
#pragma unroll
        for (int nf = 0; nf < 8; ++nf) {
            v16bf bb;
#pragma unroll
            for (int s = 0; s < 16; ++s)
                bb[s] = ldsB[(kk + (s & 8) * 2 + hf * 8 + (s & 7)) * 128 + nf * 16 + r];
            oacc[nf] = __builtin_amdgcn_wmma_f32_16x16x32_bf16(
                false, a, false, bb, (short)0, oacc[nf], false, false);
        }
    }

    // Store f32, coalesced across the 16 lanes of each half-wave.
#pragma unroll
    for (int nf = 0; nf < 8; ++nf) {
#pragma unroll
        for (int v = 0; v < 8; ++v) {
            int row = wv * 16 + hf * 8 + v;
            oBT[(size_t)row * 128 + nf * 16 + r] = oacc[nf][v];
        }
    }
}

extern "C" void kernel_launch(void* const* d_in, const int* in_sizes, int n_in,
                              void* d_out, int out_size, void* d_ws, size_t ws_size,
                              hipStream_t stream) {
    const float* x    = (const float*)d_in[0];   // [16,48,512,128]
    const float* adj  = (const float*)d_in[1];   // [16,512,512]
    const float* W    = (const float*)d_in[2];   // [128,128]
    const float* bias = (const float*)d_in[3];   // [128]
    float* out = (float*)d_out;                  // [16,48,512,128]

    dim3 grid(16 * 48 * 4);
    dim3 block(256);
    hipLaunchKernelGGL(graph_layer_fused, grid, block, 0, stream,
                       x, adj, W, bias, out);
}
